// S6Layer_32950989094981
// MI455X (gfx1250) — matrix operand order repaired and checked by reading, not compiled, not run on hardware
//
#include <hip/hip_runtime.h>
#include <hip/hip_bf16.h>
#include <math.h>

// ---------------------------------------------------------------------------
// S6 layer for MI455X (gfx1250), wave32 + WMMA bf16 path.
//   GEMMs:  v_wmma_f32_16x16x32_bf16 (fp32 accum), register-staged software
//           pipeline: global loads for tile k+1 overlap WMMAs on tile k.
//   Scan:   fused elementwise + recurrence + N-contraction, bf16 Bp/Cp streams
// ---------------------------------------------------------------------------

typedef __attribute__((ext_vector_type(16))) __bf16 v16bf;
typedef __attribute__((ext_vector_type(8)))  __bf16 v8bf;
typedef __attribute__((ext_vector_type(8)))  float  v8f;

#define S6_D_MODEL 512
#define S6_D_STATE 512
#define S6_N_DIM   16
#define S6_T       1024
#define S6_B       4
#define S6_D_INNER 1024
#define S6_KN      (S6_D_STATE * S6_N_DIM)   // 8192

#define BM 128
#define BN 64
#define BK 32

// ---------------------------------------------------------------------------
// GEMM: C[M,N] = op(A[M,K](f32) * B[K,N](f32) + bias[N])
// MODE 0: bias only.  MODE 1: bias + softplus (log1p(exp(x))).
// Block = 256 threads (8 waves). Block tile 128x64, wave tile 64x16 (4 WMMAs),
// K-step 32.  fp32->bf16 conversion during LDS store; double-buffered in
// registers so HBM latency hides under the WMMA group.
// ---------------------------------------------------------------------------
template<int MODE, typename OUTT>
__global__ __launch_bounds__(256) void gemm_bias_wmma(
    const float* __restrict__ A, const float* __restrict__ Bw,
    const float* __restrict__ bias, OUTT* __restrict__ C,
    int M, int N, int K)
{
  __shared__ __align__(64) __bf16 sA[BM][BK];   // 8 KB, row-major (M x K)
  __shared__ __align__(64) __bf16 sB[BN][BK];   // 4 KB, transposed (N x K)

  const int tid  = threadIdx.x;
  const int lane = tid & 31;
  const int wave = tid >> 5;       // 0..7
  const int wm   = wave >> 2;      // 0..1  (64-row strip)
  const int wn   = wave & 3;       // 0..3  (16-col strip)
  const int l16  = lane & 15;
  const int half = lane >> 4;      // 0/1: which K-half this lane holds
  const int bm   = blockIdx.y * BM;
  const int bn   = blockIdx.x * BN;

  // ---- per-thread staging coordinates (loop-invariant)
  int lrA[4], lcA[4];
  const float* aSrc[4];
#pragma unroll
  for (int j = 0; j < 4; ++j) {
    const int e = (tid + j * 256) * 4;          // 0..4095 (128x32 A tile)
    lrA[j] = e >> 5;
    lcA[j] = e & 31;
    aSrc[j] = A + (size_t)(bm + lrA[j]) * K + lcA[j];
  }
  int kkB[2], nnB[2];
  const float* bSrc[2];
#pragma unroll
  for (int j = 0; j < 2; ++j) {
    const int e = (tid + j * 256) * 4;          // 0..2047 (32x64 B tile)
    kkB[j] = e >> 6;
    nnB[j] = e & 63;
    bSrc[j] = Bw + (size_t)kkB[j] * N + bn + nnB[j];
  }

  // ---- prologue: stage tile k0=0 into registers
  float4 ra[4], rb[2];
#pragma unroll
  for (int j = 0; j < 4; ++j) ra[j] = *(const float4*)(aSrc[j]);
#pragma unroll
  for (int j = 0; j < 2; ++j) rb[j] = *(const float4*)(bSrc[j]);

  v8f acc[4] = {};

  for (int k0 = 0; k0 < K; k0 += BK) {
    // ---- convert + store staged tile to LDS
#pragma unroll
    for (int j = 0; j < 4; ++j) {
      sA[lrA[j]][lcA[j] + 0] = (__bf16)ra[j].x;
      sA[lrA[j]][lcA[j] + 1] = (__bf16)ra[j].y;
      sA[lrA[j]][lcA[j] + 2] = (__bf16)ra[j].z;
      sA[lrA[j]][lcA[j] + 3] = (__bf16)ra[j].w;
    }
#pragma unroll
    for (int j = 0; j < 2; ++j) {
      sB[nnB[j] + 0][kkB[j]] = (__bf16)rb[j].x;
      sB[nnB[j] + 1][kkB[j]] = (__bf16)rb[j].y;
      sB[nnB[j] + 2][kkB[j]] = (__bf16)rb[j].z;
      sB[nnB[j] + 3][kkB[j]] = (__bf16)rb[j].w;
    }
    __syncthreads();

    // ---- issue next tile's global loads now; the s_wait_loadcnt for these
    // lands at the top of the next iteration, after the WMMA group below.
    if (k0 + BK < K) {
#pragma unroll
      for (int j = 0; j < 4; ++j)
        ra[j] = *(const float4*)(aSrc[j] + (k0 + BK));
#pragma unroll
      for (int j = 0; j < 2; ++j)
        rb[j] = *(const float4*)(bSrc[j] + (size_t)(k0 + BK) * N);
      if (k0 + 2 * BK < K) {   // gfx1250 global_prefetch_b8, two tiles ahead
        __builtin_prefetch((const void*)(aSrc[0] + (k0 + 2 * BK)), 0, 1);
        __builtin_prefetch((const void*)(bSrc[0] + (size_t)(k0 + 2 * BK) * N), 0, 1);
      }
    }

    // ---- fragments: B once, all four A fragments, then WMMAs back-to-back
    const v16bf bfrag = *(const v16bf*)&sB[wn * 16 + l16][half * 16];
    v16bf afr[4];
#pragma unroll
    for (int m = 0; m < 4; ++m) {
      const int row = wm * 64 + m * 16 + l16;
      const v8bf alo = *(const v8bf*)&sA[row][half * 16];
      const v8bf ahi = *(const v8bf*)&sA[row][half * 16 + 8];
#pragma unroll
      for (int i = 0; i < 8; ++i) { afr[m][i] = alo[i]; afr[m][i + 8] = ahi[i]; }
    }
#pragma unroll
    for (int m = 0; m < 4; ++m)
      acc[m] = __builtin_amdgcn_wmma_f32_16x16x32_bf16(
          false, afr[m], false, bfrag, (short)0, acc[m], false, false);
    __syncthreads();
  }

  // ---- epilogue: C layout VGPR i -> M = i + 8*half, N = l16
  const int col = bn + wn * 16 + l16;
  const float bv = bias[col];
#pragma unroll
  for (int m = 0; m < 4; ++m) {
    const int row0 = bm + wm * 64 + m * 16 + half * 8;
#pragma unroll
    for (int i = 0; i < 8; ++i) {
      float v = acc[m][i] + bv;
      if (MODE == 1) v = (v > 20.f) ? v : log1pf(expf(v));   // softplus
      C[(size_t)(row0 + i) * N + col] = (OUTT)v;
    }
  }
}

// ---------------------------------------------------------------------------
// Fused scan: for each state (b,s,n) run h_t = dA_t h_{t-1} + v*delta*Bp, and
// contract over n with Cp to produce y_inner[b,t,s].  Block = 64 threads
// (4 s-groups x 16 n); grid = (S/4, B) = 512 blocks for WGP spread.
// ---------------------------------------------------------------------------
__global__ __launch_bounds__(64) void s6_scan_kernel(
    const float* __restrict__ delta, const __bf16* __restrict__ Bp,
    const __bf16* __restrict__ Cp, const float* __restrict__ A_log,
    float* __restrict__ y_inner, float* __restrict__ h_last)
{
  const int b = blockIdx.y;
  const int s = blockIdx.x * 4 + (threadIdx.x >> 4);
  const int n = threadIdx.x & 15;

  const float a_neg = -__expf(A_log[s * S6_N_DIM + n]);

  const float*  dp = delta + ((size_t)b * S6_T) * S6_D_STATE + s;
  const __bf16* bp = Bp + ((size_t)b * S6_T) * S6_KN + s * S6_N_DIM + n;
  const __bf16* cp = Cp + ((size_t)b * S6_T) * S6_KN + s * S6_N_DIM + n;
  float*        yp = y_inner + ((size_t)b * S6_T) * S6_D_STATE + s;

  float h = 0.f;
  for (int t = 0; t < S6_T; ++t) {
    const float dl  = dp[(size_t)t * S6_D_STATE];
    const float bv  = (float)bp[(size_t)t * S6_KN];
    const float cv  = (float)cp[(size_t)t * S6_KN];
    const float dtA = dl * a_neg;
    const float dA  = __expf(dtA);
    float v = (dtA != 0.f) ? (dA - 1.f) / dtA : 0.f;   // divide_no_nan semantics
    if (!__builtin_isfinite(v)) v = 1.f;
    h = dA * h + v * dl * bv;

    float p = h * cv;                 // contract over n (16 lanes, wave32)
    p += __shfl_xor(p, 8, 16);
    p += __shfl_xor(p, 4, 16);
    p += __shfl_xor(p, 2, 16);
    p += __shfl_xor(p, 1, 16);
    if (n == 0) yp[(size_t)t * S6_D_STATE] = p;
  }
  h_last[((size_t)b * S6_D_STATE + s) * S6_N_DIM + n] = h;
}

// ---------------------------------------------------------------------------
// RMS norm: one block per (b,t) row of 512; 256 threads x 2 elements.
// ---------------------------------------------------------------------------
__global__ __launch_bounds__(256) void s6_rmsnorm_kernel(
    const float* __restrict__ y_pre, const float* __restrict__ gamma,
    float* __restrict__ out)
{
  const int row = blockIdx.x;
  const int t   = threadIdx.x;
  const float* yr = y_pre + (size_t)row * S6_D_MODEL;
  const float v0 = yr[t];
  const float v1 = yr[t + 256];
  float ss = v0 * v0 + v1 * v1;
#pragma unroll
  for (int off = 16; off; off >>= 1) ss += __shfl_xor(ss, off, 32);

  __shared__ float red[8];
  if ((t & 31) == 0) red[t >> 5] = ss;
  __syncthreads();
  float tot = 0.f;
#pragma unroll
  for (int i = 0; i < 8; ++i) tot += red[i];

  const float r = sqrtf(tot / (float)S6_D_MODEL + 1e-8f);
  out[(size_t)row * S6_D_MODEL + t]       = gamma[t]       * v0 / r;
  out[(size_t)row * S6_D_MODEL + t + 256] = gamma[t + 256] * v1 / r;
}

// ---------------------------------------------------------------------------
extern "C" void kernel_launch(void* const* d_in, const int* in_sizes, int n_in,
                              void* d_out, int out_size, void* d_ws, size_t ws_size,
                              hipStream_t stream)
{
  const float* x       = (const float*)d_in[0];
  const float* W_in    = (const float*)d_in[1];
  const float* b_in    = (const float*)d_in[2];
  const float* W_delta = (const float*)d_in[3];
  const float* b_delta = (const float*)d_in[4];
  const float* W_B     = (const float*)d_in[5];
  const float* b_B     = (const float*)d_in[6];
  const float* W_C     = (const float*)d_in[7];
  const float* b_C     = (const float*)d_in[8];
  const float* W_out   = (const float*)d_in[9];
  const float* b_out   = (const float*)d_in[10];
  const float* gamma   = (const float*)d_in[11];
  const float* A_log   = (const float*)d_in[12];

  const int M = S6_B * S6_T;                 // 4096 rows

  // Workspace layout (fp32 first, then bf16 streams): ~168 MB total.
  float*  xp      = (float*)d_ws;                        // 4096 x 1024
  float*  deltap  = xp      + (size_t)M * S6_D_INNER;    // 4096 x 512
  float*  y_inner = deltap  + (size_t)M * S6_D_STATE;    // 4096 x 512
  float*  y_pre   = y_inner + (size_t)M * S6_D_STATE;    // 4096 x 512
  __bf16* Bp      = (__bf16*)(y_pre + (size_t)M * S6_D_MODEL);   // 4096 x 8192 bf16
  __bf16* Cp      = Bp + (size_t)M * S6_KN;                      // 4096 x 8192 bf16

  float* y_out  = (float*)d_out;
  float* h_last = y_out + (size_t)M * S6_D_MODEL;        // tuple output tail

  const dim3 blk(256);

  // 1) xp = x @ W_in + b_in
  gemm_bias_wmma<0, float><<<dim3(S6_D_INNER / BN, M / BM), blk, 0, stream>>>(
      x, W_in, b_in, xp, M, S6_D_INNER, S6_D_MODEL);
  // 2) delta = softplus(xp @ W_delta + b_delta)
  gemm_bias_wmma<1, float><<<dim3(S6_D_STATE / BN, M / BM), blk, 0, stream>>>(
      xp, W_delta, b_delta, deltap, M, S6_D_STATE, S6_D_INNER);
  // 3) Bp = xp @ W_B + b_B   (bf16 stream for the scan)
  gemm_bias_wmma<0, __bf16><<<dim3(S6_KN / BN, M / BM), blk, 0, stream>>>(
      xp, W_B, b_B, Bp, M, S6_KN, S6_D_INNER);
  // 4) Cp = xp @ W_C + b_C
  gemm_bias_wmma<0, __bf16><<<dim3(S6_KN / BN, M / BM), blk, 0, stream>>>(
      xp, W_C, b_C, Cp, M, S6_KN, S6_D_INNER);
  // 5) fused recurrence + y_inner contraction + h_last
  s6_scan_kernel<<<dim3(S6_D_STATE / 4, S6_B), dim3(64), 0, stream>>>(
      deltap, Bp, Cp, A_log, y_inner, h_last);
  // 6) y_pre = y_inner @ W_out + b_out
  gemm_bias_wmma<0, float><<<dim3(S6_D_MODEL / BN, M / BM), blk, 0, stream>>>(
      y_inner, W_out, b_out, y_pre, M, S6_D_MODEL, S6_D_STATE);
  // 7) RMS norm -> y
  s6_rmsnorm_kernel<<<dim3(M), dim3(256), 0, stream>>>(y_pre, gamma, y_out);
}
